// FeatureBilinearAttention_42949673243
// MI455X (gfx1250) — matrix operand order repaired and checked by reading, not compiled, not run on hardware
//
#include <hip/hip_runtime.h>
#include <hip/hip_bf16.h>

// Problem constants (match reference setup_inputs)
#define NN 65536
#define MM 2048
#define DD 256
#define KK 512

typedef __attribute__((ext_vector_type(16))) __bf16 bf16x16;
typedef __attribute__((ext_vector_type(8)))  float  v8f;
typedef unsigned short ushort_t;
typedef unsigned int   uint_t;
typedef unsigned int __attribute__((ext_vector_type(4))) u32x4;
typedef int          __attribute__((ext_vector_type(8))) i32x8;
typedef int          __attribute__((ext_vector_type(4))) i32x4;

// ---------- helpers ----------
__device__ __forceinline__ ushort_t f2bf(float x) {
    uint_t u = __builtin_bit_cast(uint_t, x);
    u = (u + 0x7FFFu + ((u >> 16) & 1u)) >> 16;   // round-to-nearest-even
    return (ushort_t)u;
}
__device__ __forceinline__ float bf2f(ushort_t h) {
    uint_t u = ((uint_t)h) << 16;
    return __builtin_bit_cast(float, u);
}
// native v_exp_f32 (exp2) based e^x; args in softmax are <= 0 so saturation is fine
__device__ __forceinline__ float fast_exp(float x) {
#if __has_builtin(__builtin_amdgcn_exp2f)
    return __builtin_amdgcn_exp2f(x * 1.4426950408889634f);
#else
    return __expf(x);
#endif
}
__device__ __forceinline__ float fast_tanh(float x) {
#if __has_builtin(__builtin_amdgcn_tanhf)
    return __builtin_amdgcn_tanhf(x);     // native V_TANH_F32 on gfx1250
#else
    float e = fast_exp(2.0f * x);         // inf -> t=1, 0 -> t=-1 : branchless
#if __has_builtin(__builtin_amdgcn_rcpf)
    return 1.0f - 2.0f * __builtin_amdgcn_rcpf(e + 1.0f);
#else
    return 1.0f - 2.0f / (e + 1.0f);
#endif
#endif
}
__device__ __forceinline__ float fast_rcp(float x) {
#if __has_builtin(__builtin_amdgcn_rcpf)
    return __builtin_amdgcn_rcpf(x);
#else
    return 1.0f / x;
#endif
}
// Load a 16-element bf16 fragment as two 16-byte chunks (works for LDS and global)
__device__ __forceinline__ bf16x16 ldfrag(const ushort_t* p0, const ushort_t* p1) {
    union { uint4 q[2]; bf16x16 v; } u;
    u.q[0] = *reinterpret_cast<const uint4*>(p0);
    u.q[1] = *reinterpret_cast<const uint4*>(p1);
    return u.v;
}
__device__ __forceinline__ v8f wmma_bf16(bf16x16 a, bf16x16 b, v8f c) {
    return __builtin_amdgcn_wmma_f32_16x16x32_bf16(false, a, false, b, (short)0, c, false, false);
}
__device__ __forceinline__ v8f vzero8() {
    v8f z = {0.f,0.f,0.f,0.f,0.f,0.f,0.f,0.f};
    return z;
}

// ---------- Tensor Data Mover (TDM) helpers ----------
#if __has_builtin(__builtin_amdgcn_tensor_load_to_lds)
#define USE_TDM 1
__device__ __forceinline__ void tdm_wait0() {
#if __has_builtin(__builtin_amdgcn_s_wait_tensorcnt)
    __builtin_amdgcn_s_wait_tensorcnt(0);
#else
    asm volatile("s_wait_tensorcnt 0x0" ::: "memory");
#endif
}
// 2D tile load, element size fixed at 8 bytes. Dims/strides in 8-byte units.
// D# layout per CDNA5 ISA ch.8: group0 {count=1, lds_addr, global_addr, type=2},
// group1 {data_size, tensor_dim0/1, tile_dim0/1, tensor_dim0_stride}; groups 2/3 zero (2D).
__device__ __forceinline__ void tdm_load_2d(unsigned int lds_off, const void* gptr,
                                            unsigned int tensor_d0, unsigned int tensor_d1,
                                            unsigned int tile_d0, unsigned int tile_d1,
                                            unsigned long long stride0) {
    unsigned long long ga = (unsigned long long)(__SIZE_TYPE__)gptr;
    u32x4 g0;
    g0[0] = 1u;                                              // count=1 (valid), user mode
    g0[1] = lds_off;                                         // LDS byte address
    g0[2] = (unsigned int)(ga & 0xFFFFFFFFu);                // global_addr[31:0]
    g0[3] = (unsigned int)((ga >> 32) & 0x01FFFFFFu) | (2u << 30);  // addr[56:32] | type=2
    i32x8 g1;
    g1[0] = (int)(3u << 16);                                 // data_size=3 (8 bytes)
    g1[1] = (int)((tensor_d0 & 0xFFFFu) << 16);              // tensor_dim0[15:0] @bit48
    g1[2] = (int)((tensor_d0 >> 16) | ((tensor_d1 & 0xFFFFu) << 16));
    g1[3] = (int)((tensor_d1 >> 16) | (tile_d0 << 16));      // tile_dim0 @bit112
    g1[4] = (int)(tile_d1 & 0xFFFFu);                        // tile_dim1; tile_dim2=0
    g1[5] = (int)(unsigned int)(stride0 & 0xFFFFFFFFu);      // tensor_dim0_stride[31:0]
    g1[6] = (int)(unsigned int)((stride0 >> 32) & 0xFFFFu);  // stride0[47:32]; stride1=0
    g1[7] = 0;
    i32x4 g2 = {0, 0, 0, 0};
    i32x4 g3 = {0, 0, 0, 0};
#if __clang_major__ >= 23
    i32x8 g4 = {0, 0, 0, 0, 0, 0, 0, 0};
    __builtin_amdgcn_tensor_load_to_lds(g0, g1, g2, g3, g4, 0);
#else
    __builtin_amdgcn_tensor_load_to_lds(g0, g1, g2, g3, 0);
#endif
}
#else
#define USE_TDM 0
#endif

// ---------- prep 1: UTbf[k][dd] = bf16(U[dd][k]) ----------
__global__ void prep_ut_kernel(const float* __restrict__ U, ushort_t* __restrict__ UT) {
    int idx = blockIdx.x * 256 + threadIdx.x;          // idx over K*D
    int k = idx >> 8;                                   // /256
    int dd = idx & 255;
    UT[idx] = f2bf(U[dd * KK + k]);
}

// ---------- prep 2: HdT[dcol][m] = bf16(Hd[m][dcol]) ----------
__global__ void prep_hdt_kernel(const float* __restrict__ Hd, ushort_t* __restrict__ HdT) {
    int idx = blockIdx.x * 256 + threadIdx.x;          // idx over D*M
    int dcol = idx >> 11;                               // /2048
    int m = idx & 2047;
    HdT[idx] = f2bf(Hd[m * DD + dcol]);
}

// ---------- prep 3: T[m][k] = tanh(sum_dd Hd[m][dd] * V[k][dd]) ----------
__global__ void prep_t_kernel(const float* __restrict__ Hd, const float* __restrict__ V,
                              float* __restrict__ T) {
    __shared__ float shd[16][17];
    __shared__ float sv[16][17];
    int tid = threadIdx.x;
    int ty = tid >> 4, tx = tid & 15;
    int m0 = blockIdx.x * 16;      // grid.x = M/16
    int k0 = blockIdx.y * 16;      // grid.y = K/16
    float acc = 0.f;
    for (int t = 0; t < DD / 16; ++t) {
        shd[ty][tx] = Hd[(m0 + ty) * DD + t * 16 + tx];
        sv[ty][tx]  = V [(k0 + ty) * DD + t * 16 + tx];
        __syncthreads();
#pragma unroll
        for (int e = 0; e < 16; ++e) acc += shd[ty][e] * sv[tx][e];
        __syncthreads();
    }
    T[(m0 + ty) * KK + k0 + tx] = fast_tanh(acc);
}

// ---------- prep 4: per-drug LayerNorm over K, write bf16 Dbf[m][k] ----------
__global__ void prep_ln_kernel(const float* __restrict__ T,
                               const float* __restrict__ gd, const float* __restrict__ bd,
                               ushort_t* __restrict__ Dbf) {
    __shared__ float rs[256];
    __shared__ float rq[256];
    int tid = threadIdx.x;
    int m = blockIdx.x;
    float t0 = T[m * KK + tid];
    float t1 = T[m * KK + tid + 256];
    rs[tid] = t0 + t1;
    rq[tid] = t0 * t0 + t1 * t1;
    for (int st = 128; st > 0; st >>= 1) {
        __syncthreads();
        if (tid < st) { rs[tid] += rs[tid + st]; rq[tid] += rq[tid + st]; }
    }
    __syncthreads();
    float mean = rs[0] * (1.0f / KK);
    float var  = rq[0] * (1.0f / KK) - mean * mean;
    float rstd = rsqrtf(var + 1e-5f);
    Dbf[m * KK + tid]       = f2bf((t0 - mean) * rstd * gd[tid]       + bd[tid]);
    Dbf[m * KK + tid + 256] = f2bf((t1 - mean) * rstd * gd[tid + 256] + bd[tid + 256]);
}

// ---------- fused attention kernel ----------
// 512 workgroups x 256 threads (8 waves). Each WG handles 128 RNA rows; wave w owns rows [16w,16w+16).
// Dynamic LDS layout (bytes):
//   [0,          131072) sAbf : 128x512 bf16  (A tile)
//   [131072,     196608) sB   : phase1 = Hr tile 128x256 bf16 ; phase2 = D tile 64x512 bf16
//   [196608,     229376) sHdT : 256x64 bf16
//   [229376,     245760) sP   : 128x64 bf16
//   [245760,     246784) sStat: 128x2 f32
#define SD_LDS_OFF   131072u
#define SHDT_LDS_OFF 196608u
#define ATTN_SMEM    246784

__global__ __launch_bounds__(256, 1)
void attn_kernel(const float* __restrict__ Hr,
                 const ushort_t* __restrict__ UT,
                 const ushort_t* __restrict__ Dbf,
                 const ushort_t* __restrict__ HdT,
                 const float* __restrict__ g_rna, const float* __restrict__ b_rna,
                 const float* __restrict__ q,
                 float* __restrict__ out) {
    extern __shared__ char smem[];
    ushort_t* sAbf = (ushort_t*)(smem);
    ushort_t* sB   = (ushort_t*)(smem + SD_LDS_OFF);
    ushort_t* sHdT = (ushort_t*)(smem + SHDT_LDS_OFF);
    ushort_t* sP   = (ushort_t*)(smem + 229376);
    float*    sStat= (float*)   (smem + 245760);

    const int tid  = threadIdx.x;
    const int wave = tid >> 5;
    const int lane = tid & 31;
    const int hi   = lane >> 4;     // 0 or 1 (lane half)
    const int l16  = lane & 15;
    const int row0 = blockIdx.x * 128;

    // ---- load H_rna tile -> bf16 in sB (128x256) ----
#pragma unroll
    for (int i = 0; i < 32; ++i) {
        int idx = tid + i * 256;                       // float4 index over 128*256/4
        float4 v = reinterpret_cast<const float4*>(Hr)[row0 * 64 + idx];
        uint_t lo = (uint_t)f2bf(v.x) | ((uint_t)f2bf(v.y) << 16);
        uint_t hi2 = (uint_t)f2bf(v.z) | ((uint_t)f2bf(v.w) << 16);
        uint2 pk; pk.x = lo; pk.y = hi2;
        *reinterpret_cast<uint2*>(sB + idx * 4) = pk;
    }
    __syncthreads();

    // ---- phase 1: A = tanh(Hr @ U) with running row stats ----
    float sum[8], ss[8];
#pragma unroll
    for (int j = 0; j < 8; ++j) { sum[j] = 0.f; ss[j] = 0.f; }

    const int arow = 16 * wave + l16;                  // A-fragment row for this lane
#pragma unroll 1
    for (int ctg = 0; ctg < 8; ++ctg) {                // groups of 4 col-tiles -> K=512
        v8f acc[4];
#pragma unroll
        for (int c = 0; c < 4; ++c) acc[c] = vzero8();
#pragma unroll
        for (int kk = 0; kk < 8; ++kk) {               // d=256 in chunks of 32
            const ushort_t* ap = sB + arow * 256 + kk * 32 + hi * 8;
            bf16x16 a = ldfrag(ap, ap + 16);           // one A-frag per K-chunk
#pragma unroll
            for (int c = 0; c < 4; ++c) {              // 4 independent WMMA chains
                int ct = ctg * 4 + c;
                const ushort_t* bp = UT + (ct * 16 + l16) * 256 + kk * 32 + hi * 8;
                bf16x16 b = ldfrag(bp, bp + 16);
                acc[c] = wmma_bf16(a, b, acc[c]);
            }
        }
#pragma unroll
        for (int c = 0; c < 4; ++c) {
            int col = (ctg * 4 + c) * 16 + l16;
#pragma unroll
            for (int j = 0; j < 8; ++j) {
                float t = fast_tanh(acc[c][j]);
                sum[j] += t; ss[j] += t * t;
                sAbf[(16 * wave + j + 8 * hi) * 512 + col] = f2bf(t);
            }
        }
    }
    // reduce stats over the 16 lanes of each half
#pragma unroll
    for (int j = 0; j < 8; ++j) {
#pragma unroll
        for (int msk = 1; msk < 16; msk <<= 1) {
            sum[j] += __shfl_xor(sum[j], msk);
            ss[j]  += __shfl_xor(ss[j],  msk);
        }
    }
    if (l16 == 0) {
#pragma unroll
        for (int j = 0; j < 8; ++j) {
            int r = 16 * wave + j + 8 * hi;
            float mean = sum[j] * (1.0f / KK);
            float var  = ss[j] * (1.0f / KK) - mean * mean;
            sStat[r * 2 + 0] = mean;
            sStat[r * 2 + 1] = rsqrtf(var + 1e-5f);
        }
    }
    __syncthreads();

    // ---- LN + gate pass over sAbf in place (each thread touches cols tid, tid+256 only) ----
    {
        float gq0 = g_rna[tid] * q[tid];
        float bq0 = b_rna[tid] * q[tid];
        float gq1 = g_rna[tid + 256] * q[tid + 256];
        float bq1 = b_rna[tid + 256] * q[tid + 256];
#pragma unroll 8
        for (int i = 0; i < 256; ++i) {
            int idx = i * 256 + tid;                   // over 128*512
            int r = i >> 1;
            float gq = (i & 1) ? gq1 : gq0;
            float bq = (i & 1) ? bq1 : bq0;
            float v = bf2f(sAbf[idx]);
            v = (v - sStat[r * 2]) * sStat[r * 2 + 1] * gq + bq;
            sAbf[idx] = f2bf(v);
        }
    }

    // ---- phase 2: flash loop over drugs ----
    v8f oacc[16];
#pragma unroll
    for (int i = 0; i < 16; ++i) oacc[i] = vzero8();
    float mrun[8], lrun[8];
#pragma unroll
    for (int j = 0; j < 8; ++j) { mrun[j] = -1e30f; lrun[j] = 0.f; }

    ushort_t* sD = sB;                                  // overlay
#pragma unroll 1
    for (int it = 0; it < MM / 64; ++it) {
        int m0 = it * 64;
        __syncthreads();
#if USE_TDM
        // Tensor Data Mover: one wave DMAs both tiles into LDS (TENSORcnt-tracked)
        if (wave == 0) {
            // D tile: 64 rows x 1KB contiguous = one 64KB 1D transfer (8-byte elems)
            tdm_load_2d(SD_LDS_OFF,   Dbf + m0 * 512, 8192u, 1u, 8192u, 1u, 8192ull);
            // HdT tile: 256 rows x 128B, row stride 4KB (in 8-byte elems: 16 x 256, stride 512)
            tdm_load_2d(SHDT_LDS_OFF, HdT + m0,       512u, 256u, 16u, 256u, 512ull);
            tdm_wait0();
        }
#else
        // stage D tile (64x512 bf16) and HdT tile (256x64 bf16) manually
        {
            const uint4* src = reinterpret_cast<const uint4*>(Dbf + m0 * 512);
            uint4* dst = reinterpret_cast<uint4*>(sD);
#pragma unroll
            for (int i = 0; i < 16; ++i) dst[tid + i * 256] = src[tid + i * 256];
#pragma unroll
            for (int i = 0; i < 8; ++i) {
                int idx = tid + i * 256;               // over 2048 uint4
                int r = idx >> 3, c = idx & 7;
                reinterpret_cast<uint4*>(sHdT)[idx] =
                    *reinterpret_cast<const uint4*>(HdT + r * MM + m0 + c * 8);
            }
        }
#endif
        __syncthreads();

        // prefetch next iteration's tiles toward L2/L0 while we compute
        if (it + 1 < MM / 64) {
            const char* pD = (const char*)(Dbf + (m0 + 64) * 512);
            __builtin_prefetch(pD + tid * 256, 0, 3);              // 64KB D tile
            const char* pH = (const char*)(HdT + tid * MM + (m0 + 64));
            __builtin_prefetch(pH, 0, 3);                          // 128B per HdT row
        }

        // S tile: 16 rows x 64 cols per wave (K-outer: A-frag loaded once per chunk)
        v8f sacc[4];
#pragma unroll
        for (int ct = 0; ct < 4; ++ct) sacc[ct] = vzero8();
#pragma unroll
        for (int kk = 0; kk < 16; ++kk) {              // K=512 in chunks of 32
            const ushort_t* ap = sAbf + arow * 512 + kk * 32 + hi * 8;
            bf16x16 a = ldfrag(ap, ap + 16);
#pragma unroll
            for (int ct = 0; ct < 4; ++ct) {           // 4 independent WMMA chains
                const ushort_t* bp = sD + (ct * 16 + l16) * 512 + kk * 32 + hi * 8;
                bf16x16 b = ldfrag(bp, bp + 16);
                sacc[ct] = wmma_bf16(a, b, sacc[ct]);
            }
        }

        // online softmax (rows distributed as (j, half); cols over 16 lanes)
        float mx[8];
#pragma unroll
        for (int j = 0; j < 8; ++j) {
            float m4 = fmaxf(fmaxf(sacc[0][j], sacc[1][j]), fmaxf(sacc[2][j], sacc[3][j]));
#pragma unroll
            for (int msk = 1; msk < 16; msk <<= 1) m4 = fmaxf(m4, __shfl_xor(m4, msk));
            mx[j] = m4;
        }
        float scale[8], rs[8];
#pragma unroll
        for (int j = 0; j < 8; ++j) {
            float mnew = fmaxf(mrun[j], mx[j]);
            scale[j] = fast_exp(mrun[j] - mnew);
            mrun[j] = mnew;
            rs[j] = 0.f;
        }
#pragma unroll
        for (int ct = 0; ct < 4; ++ct) {
#pragma unroll
            for (int j = 0; j < 8; ++j) {
                float p = fast_exp(sacc[ct][j] - mrun[j]);
                rs[j] += p;
                sP[(16 * wave + j + 8 * hi) * 64 + ct * 16 + l16] = f2bf(p);
            }
        }
#pragma unroll
        for (int j = 0; j < 8; ++j) {
#pragma unroll
            for (int msk = 1; msk < 16; msk <<= 1) rs[j] += __shfl_xor(rs[j], msk);
            lrun[j] = lrun[j] * scale[j] + rs[j];
        }
        // rescale O accumulators
#pragma unroll
        for (int dct = 0; dct < 16; ++dct)
#pragma unroll
            for (int j = 0; j < 8; ++j) oacc[dct][j] *= scale[j];

        // O += P @ Hd_tile (K-dim = 64 drugs -> 2 chunks of 32; A-frag reused across dct)
#pragma unroll
        for (int kk2 = 0; kk2 < 2; ++kk2) {
            const ushort_t* ap = sP + arow * 64 + kk2 * 32 + hi * 8;
            bf16x16 a = ldfrag(ap, ap + 16);
#pragma unroll
            for (int dct = 0; dct < 16; ++dct) {       // 16 independent WMMA chains
                const ushort_t* bp = sHdT + (dct * 16 + l16) * 64 + kk2 * 32 + hi * 8;
                bf16x16 b = ldfrag(bp, bp + 16);
                oacc[dct] = wmma_bf16(a, b, oacc[dct]);
            }
        }
    }

    // ---- epilogue: normalize (8 reciprocals, not 128 divides) and store ----
    float rl[8];
#pragma unroll
    for (int j = 0; j < 8; ++j) rl[j] = fast_rcp(lrun[j]);
#pragma unroll
    for (int dct = 0; dct < 16; ++dct) {
#pragma unroll
        for (int j = 0; j < 8; ++j) {
            int r = row0 + 16 * wave + j + 8 * hi;
            int c = dct * 16 + l16;
            out[r * DD + c] = oacc[dct][j] * rl[j];
        }
    }
}

// ---------- host launcher ----------
extern "C" void kernel_launch(void* const* d_in, const int* in_sizes, int n_in,
                              void* d_out, int out_size, void* d_ws, size_t ws_size,
                              hipStream_t stream) {
    (void)in_sizes; (void)n_in; (void)out_size; (void)ws_size;
    const float* Hr     = (const float*)d_in[0];
    const float* Hd     = (const float*)d_in[1];
    const float* U      = (const float*)d_in[2];
    const float* V      = (const float*)d_in[3];
    const float* q      = (const float*)d_in[4];
    const float* g_rna  = (const float*)d_in[5];
    const float* b_rna  = (const float*)d_in[6];
    const float* g_drug = (const float*)d_in[7];
    const float* b_drug = (const float*)d_in[8];
    float* out = (float*)d_out;

    char* ws = (char*)d_ws;
    ushort_t* UT   = (ushort_t*)(ws);                         // 512*256*2  = 256 KB
    ushort_t* HdT  = (ushort_t*)(ws + 262144);                // 256*2048*2 = 1 MB
    ushort_t* Dbf  = (ushort_t*)(ws + 262144 + 1048576);      // 2048*512*2 = 2 MB
    float*    Tf32 = (float*)   (ws + 262144 + 1048576 + 2097152); // 2048*512*4 = 4 MB

    prep_ut_kernel <<<(KK * DD) / 256, 256, 0, stream>>>(U, UT);
    prep_hdt_kernel<<<(DD * MM) / 256, 256, 0, stream>>>(Hd, HdT);
    dim3 gT(MM / 16, KK / 16);
    prep_t_kernel  <<<gT, 256, 0, stream>>>(Hd, V, Tf32);
    prep_ln_kernel <<<MM, 256, 0, stream>>>(Tf32, g_drug, b_drug, Dbf);
    attn_kernel    <<<NN / 128, 256, ATTN_SMEM, stream>>>(Hr, UT, Dbf, HdT,
                                                          g_rna, b_rna, q, out);
}
